// MultiHeadAttention_18210661335451
// MI455X (gfx1250) — matrix operand order repaired and checked by reading, not compiled, not run on hardware
//
#include <hip/hip_runtime.h>
#include <hip/hip_bf16.h>

// MI455X (gfx1250) fused multi-head attention, f16 WMMA with f32 accumulate.
typedef __attribute__((ext_vector_type(16))) _Float16 v16h;
typedef __attribute__((ext_vector_type(8)))  float    v8f;

constexpr int kHeads  = 16;
constexpr int kDModel = 1024;
constexpr int kDepth  = 64;
constexpr int kB      = 4;
constexpr int kS      = 1024;
constexpr int kSm1    = 1023;          // S-1
constexpr int kMRows  = kB * kSm1;     // 4092 rows for the two big GEMMs

// ---- CDNA5 WMMA 16-bit operand index maps (ISA 7.12.2) ----
// A-matrix 16x32 f16: lane%16 = M row; slot s: s<8 -> K = s + 8*(lane/16),
// s>=8 -> K = 16 + (s-8) + 8*(lane/16).  => two contiguous 8-element runs.
__device__ __forceinline__ int aK(int s, int g) {
  return ((s < 8) ? s : (8 + s)) + (g << 3);
}
// B-matrix 32x16 f16: lane%16 = N col; same K map as A (two 8-element runs)
__device__ __forceinline__ int bK(int s, int g) { return ((s < 8) ? s : (8 + s)) + (g << 3); }

__device__ __forceinline__ v8f wmma16(v16h a, v16h b, v8f c) {
  // (neg_a, A, neg_b, B, c_mod, C, reuse_a, reuse_b)
  return __builtin_amdgcn_wmma_f32_16x16x32_f16(false, a, false, b, (short)0, c,
                                                false, false);
}

// Contiguous A-operand load (f32 source), slots map to p0[0..7], p0[16..23].
__device__ __forceinline__ v16h loadA_f32(const float* __restrict__ p0) {
  v16h a;
#pragma unroll
  for (int s = 0; s < 8; ++s) {
    a[s]     = (_Float16)p0[s];
    a[s + 8] = (_Float16)p0[16 + s];
  }
  return a;
}

// =====================================================================
// Kernel 1: vv = v @ Wv + bv  -> stored head-split as f16: vh16[b][h][s][d]
// M=4092, N=1024, K=1024. Each wave: 16x64 stripe (4 C tiles, shared A).
// =====================================================================
__global__ __launch_bounds__(256) void k_gemm_vv(
    const float* __restrict__ v, const float* __restrict__ Wv,
    const float* __restrict__ bv, _Float16* __restrict__ vh16) {
  const int lane = threadIdx.x & 31, wave = threadIdx.x >> 5;
  const int g = lane >> 4, ml = lane & 15;
  const int tile = blockIdx.x * 8 + wave;        // 4096 wave-stripes
  const int mt = tile >> 4;                      // 256 M tiles
  const int ng = tile & 15;                      // 16 N groups of 64
  const int m = mt * 16 + ml;
  const int mc = (m < kMRows) ? m : (kMRows - 1);
  const float* arow = v + (size_t)mc * kDModel;

  v8f acc[4] = {};
  for (int k0 = 0; k0 < kDModel; k0 += 32) {
    __builtin_prefetch(arow + k0 + 32, 0, 1);    // speculative; drop-safe
    v16h a = loadA_f32(arow + k0 + (g << 3));
#pragma unroll
    for (int c = 0; c < 4; ++c) {
      const int n = ng * 64 + c * 16 + ml;
      v16h bm;
#pragma unroll
      for (int s = 0; s < 16; ++s)
        bm[s] = (_Float16)Wv[(size_t)(k0 + bK(s, g)) * kDModel + n];
      acc[c] = wmma16(a, bm, acc[c]);
    }
  }
  const int h = ng;                              // 64-wide group == one head
#pragma unroll
  for (int jv = 0; jv < 8; ++jv) {
    int mm = mt * 16 + jv + (g << 3);
    if (mm < kMRows) {
      int bb = mm / kSm1, srow = mm % kSm1;      // hoisted out of c loop
      size_t rbase = (((size_t)(bb * kHeads + h) * kSm1) + srow) * kDepth;
#pragma unroll
      for (int c = 0; c < 4; ++c) {
        int n = ng * 64 + c * 16 + ml;
        vh16[rbase + c * 16 + ml] = (_Float16)(acc[c][jv] + bv[n]);
      }
    }
  }
}

// =====================================================================
// Kernel 2 (fused): scores = qh.khT (sliced [1:, :-1]), relu(x/8) + mask*-1e9,
// row softmax, write normalized att. One block = one (b,h,16-query) stripe.
// Each wave owns 128 key columns (8 C tiles) kept in registers; cross-wave
// max/sum reduction via tiny LDS. att is written exactly once.
// =====================================================================
__global__ __launch_bounds__(256) void k_attn(
    const float* __restrict__ q, const float* __restrict__ kmat,
    const float* __restrict__ mask, float* __restrict__ att) {
  const int tid = threadIdx.x;
  const int lane = tid & 31, wave = tid >> 5;
  const int g = lane >> 4, ml = lane & 15;
  const int qt = blockIdx.x & 63;                // 64 query tiles
  const int bh = blockIdx.x >> 6;                // 0..63
  const int b = bh >> 4, h = bh & 15;

  __shared__ float wred[8][16];
  __shared__ float rowmax[16];
  __shared__ float rowsum[16];

  // A operand (queries): row = qt*16+ml, global s = 1+row; reused for all tiles
  v16h a0, a1;
  {
    int m = qt * 16 + ml;
    int qc = (m < kSm1) ? m : (kSm1 - 1);
    const float* p0 =
        q + ((size_t)(b * kS + 1 + qc)) * kDModel + h * kDepth + (g << 3);
#pragma unroll
    for (int s = 0; s < 8; ++s) {
      a0[s] = (_Float16)p0[s];      a0[s + 8] = (_Float16)p0[16 + s];
      a1[s] = (_Float16)p0[32 + s]; a1[s + 8] = (_Float16)p0[48 + s];
    }
  }

  // 8 key tiles per wave: columns j = (wave*8+t)*16 + ml
  v8f acc[8];
#pragma unroll
  for (int t = 0; t < 8; ++t) {
    int j = (wave * 8 + t) * 16 + ml;
    int jc = (j < kSm1) ? j : (kSm1 - 1);
    // B slots match A: kp[0..7], kp[16..23] per half-wave, +32 for b1
    const float* kp =
        kmat + ((size_t)(b * kS + jc)) * kDModel + h * kDepth + (g << 3);
    v16h b0, b1;
#pragma unroll
    for (int s = 0; s < 8; ++s) {
      b0[s] = (_Float16)kp[s];      b0[s + 8] = (_Float16)kp[16 + s];
      b1[s] = (_Float16)kp[32 + s]; b1[s + 8] = (_Float16)kp[48 + s];
    }
    v8f c = {};
    c = wmma16(a0, b0, c);
    c = wmma16(a1, b1, c);
    acc[t] = c;
  }

  // relu(x/sqrt(64)) + mask*-1e9 ; invalid cols -> -3e38 so exp()->0
#pragma unroll
  for (int t = 0; t < 8; ++t) {
    int j = (wave * 8 + t) * 16 + ml;
    int jc = (j < kSm1) ? j : (kSm1 - 1);
#pragma unroll
    for (int jv = 0; jv < 8; ++jv) {
      int qi = qt * 16 + jv + (g << 3);
      int qc = (qi < kSm1) ? qi : (kSm1 - 1);
      float x = fmaxf(acc[t][jv] * 0.125f, 0.0f);
      x += mask[((size_t)b * kSm1 + qc) * kSm1 + jc] * (-1.0e9f);
      acc[t][jv] = (j < kSm1) ? x : -3.0e38f;
    }
  }

  // ---- row max: in-wave shuffle over 16 column-lanes, then cross-wave ----
#pragma unroll
  for (int jv = 0; jv < 8; ++jv) {
    float mx = acc[0][jv];
#pragma unroll
    for (int t = 1; t < 8; ++t) mx = fmaxf(mx, acc[t][jv]);
#pragma unroll
    for (int off = 1; off < 16; off <<= 1)
      mx = fmaxf(mx, __shfl_xor(mx, off, 32));
    if (ml == 0) wred[wave][jv + (g << 3)] = mx;
  }
  __syncthreads();
  if (tid < 16) {
    float mx = wred[0][tid];
    for (int w = 1; w < 8; ++w) mx = fmaxf(mx, wred[w][tid]);
    rowmax[tid] = mx;
  }
  __syncthreads();

  // ---- exp and row sum ----
#pragma unroll
  for (int jv = 0; jv < 8; ++jv) {
    float rm = rowmax[jv + (g << 3)];
    float sm = 0.0f;
#pragma unroll
    for (int t = 0; t < 8; ++t) {
      float e = __expf(acc[t][jv] - rm);
      acc[t][jv] = e;
      sm += e;
    }
#pragma unroll
    for (int off = 1; off < 16; off <<= 1) sm += __shfl_xor(sm, off, 32);
    if (ml == 0) wred[wave][jv + (g << 3)] = sm;
  }
  __syncthreads();
  if (tid < 16) {
    float sm = 0.0f;
    for (int w = 0; w < 8; ++w) sm += wred[w][tid];
    rowsum[tid] = sm;
  }
  __syncthreads();

  // ---- normalize and write att (single pass over the 268MB output) ----
#pragma unroll
  for (int jv = 0; jv < 8; ++jv) {
    int qi = qt * 16 + jv + (g << 3);
    if (qi < kSm1) {
      float inv = 1.0f / rowsum[jv + (g << 3)];
      size_t base = ((size_t)bh * kSm1 + qi) * kSm1;
#pragma unroll
      for (int t = 0; t < 8; ++t) {
        int j = (wave * 8 + t) * 16 + ml;
        if (j < kSm1) att[base + j] = acc[t][jv] * inv;
      }
    }
  }
}

// =====================================================================
// Kernel 3: ctx = att @ vh  per (b,h): M=1023, N=64, K=1023.
// Each wave: one 16-row stripe across all 64 cols (4 tiles, shared A).
// ctx stored merged-head f16: ctx16[b*1023+q][h*64+d]
// =====================================================================
__global__ __launch_bounds__(256) void k_ctx(
    const float* __restrict__ att, const _Float16* __restrict__ vh16,
    _Float16* __restrict__ ctx16) {
  const int lane = threadIdx.x & 31, wave = threadIdx.x >> 5;
  const int g = lane >> 4, ml = lane & 15;
  const int tile = blockIdx.x * 8 + wave;        // 4096 wave-stripes
  const int mt = tile & 63;                      // 64 M tiles
  const int bh = tile >> 6;                      // 64 (b,h)
  const int b = bh >> 4, h = bh & 15;
  const int m = mt * 16 + ml;
  const int mc = (m < kSm1) ? m : (kSm1 - 1);
  const float* arow = att + ((size_t)bh * kSm1 + mc) * kSm1;
  const _Float16* vbase = vh16 + (size_t)bh * kSm1 * kDepth;

  v8f acc[4] = {};
  for (int k0 = 0; k0 < kSm1; k0 += 32) {        // 32 steps, last partial
    __builtin_prefetch(arow + k0 + 32, 0, 1);
    v16h a;
    if (k0 + 32 <= kSm1) {                       // uniform: full fast path
      a = loadA_f32(arow + k0 + (g << 3));
    } else {                                     // final partial k-step only
#pragma unroll
      for (int s = 0; s < 16; ++s) {
        int ka = k0 + aK(s, g);
        int kac = (ka < kSm1) ? ka : (kSm1 - 1);
        float av = arow[kac];
        a[s] = (_Float16)((ka < kSm1) ? av : 0.0f);  // zero kills k=1023 term
      }
    }
#pragma unroll
    for (int c = 0; c < 4; ++c) {
      const int n = c * 16 + ml;
      v16h bm;
#pragma unroll
      for (int s = 0; s < 16; ++s) {
        int kb = k0 + bK(s, g);
        int kbc = (kb < kSm1) ? kb : (kSm1 - 1);
        bm[s] = vbase[(size_t)kbc * kDepth + n];
      }
      acc[c] = wmma16(a, bm, acc[c]);
    }
  }
#pragma unroll
  for (int jv = 0; jv < 8; ++jv) {
    int mm = mt * 16 + jv + (g << 3);
    if (mm < kSm1) {
      size_t rbase = ((size_t)(b * kSm1 + mm)) * kDModel + h * kDepth;
#pragma unroll
      for (int c = 0; c < 4; ++c)
        ctx16[rbase + c * 16 + ml] = (_Float16)acc[c][jv];
    }
  }
}

// =====================================================================
// Kernel 4: output = ctx @ Wd + bd  (M=4092, N=1024, K=1024), f32 out.
// Each wave: 16x64 stripe (4 C tiles, shared A). A is f16: b128 loads.
// =====================================================================
__global__ __launch_bounds__(256) void k_gemm_out(
    const _Float16* __restrict__ ctx16, const float* __restrict__ Wd,
    const float* __restrict__ bd, float* __restrict__ out) {
  const int lane = threadIdx.x & 31, wave = threadIdx.x >> 5;
  const int g = lane >> 4, ml = lane & 15;
  const int tile = blockIdx.x * 8 + wave;        // 4096 wave-stripes
  const int mt = tile >> 4;                      // 256 M tiles
  const int ng = tile & 15;                      // 16 N groups of 64
  const int m = mt * 16 + ml;
  const int mc = (m < kMRows) ? m : (kMRows - 1);
  const _Float16* arow = ctx16 + (size_t)mc * kDModel;

  v8f acc[4] = {};
  for (int k0 = 0; k0 < kDModel; k0 += 32) {
    __builtin_prefetch(arow + k0 + 32, 0, 1);
    v16h a;
    {
      const _Float16* p0 = arow + k0 + (g << 3);
#pragma unroll
      for (int s = 0; s < 8; ++s) {
        a[s]     = p0[s];
        a[s + 8] = p0[16 + s];
      }
    }
#pragma unroll
    for (int c = 0; c < 4; ++c) {
      const int n = ng * 64 + c * 16 + ml;
      v16h bm;
#pragma unroll
      for (int s = 0; s < 16; ++s)
        bm[s] = (_Float16)Wd[(size_t)(k0 + bK(s, g)) * kDModel + n];
      acc[c] = wmma16(a, bm, acc[c]);
    }
  }
#pragma unroll
  for (int jv = 0; jv < 8; ++jv) {
    int mm = mt * 16 + jv + (g << 3);
    if (mm < kMRows) {
      size_t rbase = (size_t)mm * kDModel + ng * 64;
#pragma unroll
      for (int c = 0; c < 4; ++c) {
        int n = ng * 64 + c * 16 + ml;
        out[rbase + c * 16 + ml] = acc[c][jv] + bd[n];
      }
    }
  }
}

extern "C" void kernel_launch(void* const* d_in, const int* in_sizes, int n_in,
                              void* d_out, int out_size, void* d_ws,
                              size_t ws_size, hipStream_t stream) {
  (void)in_sizes; (void)n_in; (void)out_size; (void)ws_size;
  const float* v    = (const float*)d_in[0];
  const float* kk   = (const float*)d_in[1];
  const float* q    = (const float*)d_in[2];
  const float* mask = (const float*)d_in[3];
  const float* Wv   = (const float*)d_in[4];
  const float* bv   = (const float*)d_in[5];
  const float* Wd   = (const float*)d_in[6];
  const float* bd   = (const float*)d_in[7];

  float* out = (float*)d_out;
  float* att = out + (size_t)kB * kSm1 * kDModel;     // att region of d_out

  const size_t vh_bytes = (size_t)kB * kHeads * kSm1 * kDepth * sizeof(_Float16);
  _Float16* vh16  = (_Float16*)d_ws;
  _Float16* ctx16 = (_Float16*)((char*)d_ws + vh_bytes);

  k_gemm_vv <<<512,  256, 0, stream>>>(v, Wv, bv, vh16);
  k_attn    <<<4096, 256, 0, stream>>>(q, kk, mask, att);
  k_ctx     <<<512,  256, 0, stream>>>(att, vh16, ctx16);
  k_gemm_out<<<512,  256, 0, stream>>>(ctx16, Wd, bd, out);
}
